// DiCoDiGS_16647293239710
// MI455X (gfx1250) — matrix operand order repaired and checked by reading, not compiled, not run on hardware
//
#include <hip/hip_runtime.h>
#include <hip/hip_bf16.h>
#include <math.h>

// ---------------- fixed problem dims (from reference setup_inputs) ---------
#define B_    8
#define CIN   4
#define T_    64000
#define CH    64          // backbone channels
#define KSPK  4           // MAX_SPK
#define NBIN  36
#define KDIM  320         // 64 in-ch * 5 taps for conv2 GEMM
#define TT1   128         // time tile for fused conv kernel
#define NT1   (T_/TT1)    // 500
#define TT3   512         // time tile for refine kernel
#define NT3   (T_/TT3)    // 125

// output layout: out (8*4*64000), doa_logits (1152), refined_doa (1152)
#define OUT1  (B_*KSPK*T_)          // 2,048,000
#define OUT2  (OUT1 + B_*KSPK*NBIN) // +1152

// workspace layout (float offsets)
#define WS_POOL   0        // [8*64]  sum_t h
#define WS_SAB    512      // [8*4]   sum_t |mixture|
#define WS_RSUM   544      // [8*4]   sum_t refined out
#define WS_CONST  576      // [8*4*64] folded refine-conv constants
#define WS_COARSE 4096     // [8*4*64000] coarse separation (refine input)

typedef _Float16 half_t;
typedef __attribute__((ext_vector_type(16))) _Float16 v16h;
typedef __attribute__((ext_vector_type(8)))  float    v8f;

// load 16 contiguous f16 from LDS (two b128 loads)
__device__ inline v16h ld_frag16(const half_t* p) {
  union { uint4 u[2]; v16h v; } t;
  t.u[0] = *(const uint4*)(p);
  t.u[1] = *(const uint4*)(p + 8);
  return t.v;
}
// load two 8-half runs (A fragment: k-runs [base, base+8) and [base+16, base+24))
__device__ inline v16h ld_frag8x2(const half_t* p) {
  union { uint4 u[2]; v16h v; } t;
  t.u[0] = *(const uint4*)(p);
  t.u[1] = *(const uint4*)(p + 16);
  return t.v;
}

// ---------------------------------------------------------------------------
// K0: zero the small accumulator region of the workspace
// ---------------------------------------------------------------------------
__global__ void k0_init(float* ws) {
  for (int i = threadIdx.x; i < WS_CONST; i += blockDim.x) ws[i] = 0.f;
}

// ---------------------------------------------------------------------------
// K1: fused conv1(4->64,k5)+relu -> conv2(64->64,k5,WMMA)+relu -> coarse(64->4)
//     + pooled partial sums + sum|mixture| partial sums.
//     One block handles one (batch, 128-sample time tile).
// ---------------------------------------------------------------------------
__global__ void __launch_bounds__(256)
k1_backbone(const float* __restrict__ mix,
            const float* __restrict__ w1, const float* __restrict__ b1,
            const float* __restrict__ w2, const float* __restrict__ b2,
            const float* __restrict__ sepw, const float* __restrict__ sepb,
            float* __restrict__ ws) {
  extern __shared__ __align__(16) char smem[];
  half_t* wA    = (half_t*)(smem);              // [64][320] f16 conv2 weights
  half_t* h1s   = (half_t*)(smem + 40960);      // [64][136] f16 conv1 output tile
  float*  mixs  = (float*) (smem + 58368);      // [4][136]  mixture tile
  float*  w1s   = (float*) (smem + 60544);      // [64*4*5]  conv1 weights
  float*  sab   = (float*) (smem + 65664);      // [4]       |mix| partials
  half_t* bcol  = (half_t*)(smem + 65696);      // [128][320] im2col (B matrix)
  float*  hst   = (float*) (smem + 65696);      // [64][132] aliases bcol (phase 2)

  const int tid = threadIdx.x;
  const int b   = blockIdx.x / NT1;
  const int t0  = (blockIdx.x % NT1) * TT1;

  // stage conv2 weights as f16 (linear copy: k = c*5+tap matches bb_w2 layout)
  for (int i = tid; i < CH * KDIM; i += 256) wA[i] = (half_t)w2[i];
  for (int i = tid; i < CH * CIN * 5; i += 256) w1s[i] = w1[i];
  // stage mixture tile with halo 4 (conv1 halo 2 + conv2 halo 2)
  for (int i = tid; i < CIN * 136; i += 256) {
    int c = i / 136, ii = i % 136;
    int gt = t0 - 4 + ii;
    mixs[c * 136 + ii] =
        (gt >= 0 && gt < T_) ? mix[(size_t)(b * CIN + c) * T_ + gt] : 0.f;
  }
  if (tid < 4) {
    sab[tid] = 0.f;
    // hint: pull next time tile toward the caches (global_prefetch_b8)
    int nt = t0 + TT1;
    if (nt < T_) __builtin_prefetch(mix + (size_t)(b * CIN + tid) * T_ + nt, 0, 1);
  }
  __syncthreads();

  // |mixture| partial sums over owned region [t0, t0+128)
  if (tid < TT1) {
    #pragma unroll
    for (int c = 0; c < CIN; ++c)
      atomicAdd(&sab[c], fabsf(mixs[c * 136 + 4 + tid]));
  }

  // conv1 + relu into h1s (local idx ii <-> global t0-2+ii)
  for (int i = tid; i < CH * 132; i += 256) {
    int c = i / 132, ii = i % 132;
    int th = t0 - 2 + ii;
    float acc = 0.f;
    if (th >= 0 && th < T_) {
      acc = b1[c];
      #pragma unroll
      for (int ci = 0; ci < CIN; ++ci)
        #pragma unroll
        for (int tp = 0; tp < 5; ++tp)
          acc += w1s[(c * CIN + ci) * 5 + tp] * mixs[ci * 136 + ii + tp];
      acc = fmaxf(acc, 0.f);
    }
    h1s[c * 136 + ii] = (half_t)acc;
  }
  __syncthreads();

  // im2col: bcol[t][k] (column-major in k so fragments are contiguous runs)
  for (int i = tid; i < TT1 * KDIM; i += 256) {
    int t = i / KDIM, k = i % KDIM;
    int c = k / 5, tp = k % 5;
    bcol[i] = h1s[c * 136 + t + tp];
  }
  __syncthreads();

  if (tid < 4) atomicAdd(&ws[WS_SAB + b * 4 + tid], sab[tid]);

  // ---- WMMA GEMM: D(64 x 128) = W(64 x 320) x B(320 x 128), f16 -> f32 ----
  const int wv = tid >> 5;          // wave id = column tile (8 x 16 cols)
  const int l  = tid & 31;
  const int nlo = l & 15;
  const int hi  = l >> 4;
  const int col = wv * 16 + nlo;

  v8f acc[4];
  #pragma unroll
  for (int mt = 0; mt < 4; ++mt) acc[mt] = (v8f){0, 0, 0, 0, 0, 0, 0, 0};

  #pragma unroll
  for (int kk = 0; kk < 10; ++kk) {
    // B fragment: lanes 0-15 K[kk*32 .. +16), lanes 16-31 K[kk*32+16 .. +32)
    v16h bf = ld_frag16(bcol + col * KDIM + kk * 32 + hi * 16);
    #pragma unroll
    for (int mt = 0; mt < 4; ++mt) {
      int m = mt * 16 + nlo;
      // A fragment: contiguous run [hi*16, hi*16+16)
      v16h af = ld_frag16(wA + m * KDIM + kk * 32 + hi * 16);
      acc[mt] = __builtin_amdgcn_wmma_f32_16x16x32_f16(
          false, af, false, bf, (short)0, acc[mt], false, false);
    }
  }
  __syncthreads();   // everyone done reading bcol before aliasing as hst

  // bias + relu, stage h tile to LDS
  #pragma unroll
  for (int mt = 0; mt < 4; ++mt)
    #pragma unroll
    for (int j = 0; j < 8; ++j) {
      int m = mt * 16 + j + hi * 8;
      hst[m * 132 + col] = fmaxf(acc[mt][j] + b2[m], 0.f);
    }
  __syncthreads();

  // coarse = sep_w . h  (initial refine state), stored to workspace
  if (tid < TT1) {
    int gt = t0 + tid;
    #pragma unroll
    for (int kk = 0; kk < KSPK; ++kk) {
      float s = sepb[kk];
      for (int m = 0; m < CH; ++m) s += sepw[kk * CH + m] * hst[m * 132 + tid];
      ws[WS_COARSE + (size_t)(b * KSPK + kk) * T_ + gt] = s;
    }
  }
  // pooled partial: sum over this tile's columns
  if (tid < CH) {
    float s = 0.f;
    for (int c = 0; c < TT1; ++c) s += hst[tid * 132 + c];
    atomicAdd(&ws[WS_POOL + b * CH + tid], s);
  }
}

// ---------------------------------------------------------------------------
// K2: DOA head (logits -> d_out, argmax -> emb gather), sp_cond, and the
//     folded refine-conv constants const[b][k][c] (32 constant channels).
// ---------------------------------------------------------------------------
__global__ void __launch_bounds__(256)
k2_head(const float* __restrict__ doaw, const float* __restrict__ doab,
        const float* __restrict__ embp, const float* __restrict__ spw,
        const float* __restrict__ spb,  const float* __restrict__ rw1,
        const float* __restrict__ rb1,  float* __restrict__ ws,
        float* __restrict__ dout) {
  __shared__ float s_pool[B_ * CH];
  __shared__ float s_sab[B_ * CIN];
  __shared__ float s_log[B_ * KSPK * NBIN];
  __shared__ float s_sp[B_ * 16];
  __shared__ float s_dc[B_ * KSPK * 16];
  const int tid = threadIdx.x;
  const float invT = 1.f / (float)T_;

  for (int i = tid; i < B_ * CH; i += 256) s_pool[i] = ws[WS_POOL + i] * invT;
  if (tid < B_ * CIN) s_sab[tid] = ws[WS_SAB + tid] * invT;
  __syncthreads();

  for (int i = tid; i < B_ * KSPK * NBIN; i += 256) {
    int b = i / (KSPK * NBIN), r = i % (KSPK * NBIN);
    float v = doab[r];
    for (int c = 0; c < CH; ++c) v += doaw[r * CH + c] * s_pool[b * CH + c];
    s_log[i] = v;
    dout[OUT1 + i] = v;
  }
  if (tid < B_ * 16) {
    int b = tid / 16, j = tid % 16;
    float v = spb[j];
    #pragma unroll
    for (int c = 0; c < CIN; ++c) v += spw[j * CIN + c] * s_sab[b * CIN + c];
    s_sp[tid] = v;
  }
  __syncthreads();

  if (tid < B_ * KSPK) {
    const float* lg = &s_log[tid * NBIN];
    int best = 0; float bv = lg[0];
    for (int j = 1; j < NBIN; ++j) if (lg[j] > bv) { bv = lg[j]; best = j; }
    for (int j = 0; j < 16; ++j) s_dc[tid * 16 + j] = embp[best * 16 + j];
  }
  __syncthreads();

  // const[bk][c] = rb1[c] + sum_tap sum_j w1[c][2+j][tap]*dc + w1[c][18+j][tap]*sp
  for (int i = tid; i < B_ * KSPK * CH; i += 256) {
    int bk = i / CH, c = i % CH, b = bk / KSPK;
    float v = rb1[c];
    #pragma unroll
    for (int tp = 0; tp < 5; ++tp) {
      const float* wc = rw1 + c * 170 + tp;
      for (int j = 0; j < 16; ++j)
        v += wc[(2 + j) * 5] * s_dc[bk * 16 + j] + wc[(18 + j) * 5] * s_sp[b * 16 + j];
    }
    ws[WS_CONST + i] = v;
  }
}

// ---------------------------------------------------------------------------
// K3: all 3 refine steps fused in LDS. Only channels {out, ref} vary in time;
//     the 32 conditioning channels are folded into cst[c]. Halo 9 = 3 per step.
//     One block handles one (b,k, 512-sample tile).
// ---------------------------------------------------------------------------
__global__ void __launch_bounds__(256)
k3_refine(const float* __restrict__ mix, const float* __restrict__ rw1,
          const float* __restrict__ rw2, const float* __restrict__ rb2,
          float* __restrict__ dout, float* __restrict__ ws) {
  extern __shared__ __align__(16) char smem3[];
  float* bufA = (float*)(smem3);            // [530]
  float* bufB = (float*)(smem3 + 2128);     // [530]
  float* refs = (float*)(smem3 + 4256);     // [530]
  float* w1o  = (float*)(smem3 + 6384);     // [64*5] taps vs out channel
  float* w1r  = (float*)(smem3 + 7664);     // [64*5] taps vs ref channel
  float* w2s  = (float*)(smem3 + 8944);     // [64*3]
  float* cst  = (float*)(smem3 + 9712);     // [64]
  float* red  = (float*)(smem3 + 9968);     // [1]
  float* ybuf = (float*)(smem3 + 9984);     // [64][528] relu(inner conv)

  const int tid  = threadIdx.x;
  const int bk   = blockIdx.x / NT3;
  const int t0   = (blockIdx.x % NT3) * TT3;
  const int b    = bk >> 2;

  for (int i = tid; i < 530; i += 256) {
    int gt = t0 - 9 + i;
    bool ok = (gt >= 0 && gt < T_);
    bufA[i] = ok ? ws[WS_COARSE + (size_t)bk * T_ + gt] : 0.f;
    refs[i] = ok ? mix[(size_t)(b * CIN) * T_ + gt] : 0.f;  // channel 0
  }
  for (int i = tid; i < CH * 5; i += 256) {
    int c = i / 5, tp = i % 5;
    w1o[i] = rw1[c * 170 + tp];        // in-channel 0
    w1r[i] = rw1[c * 170 + 5 + tp];    // in-channel 1
  }
  for (int i = tid; i < CH * 3; i += 256) w2s[i] = rw2[i];
  if (tid < CH) cst[tid] = ws[WS_CONST + bk * CH + tid];
  if (tid == 0) red[0] = 0.f;
  const float bias2 = rb2[0];
  __syncthreads();

  float* cur = bufA;
  float* nxt = bufB;
  for (int s = 0; s < 3; ++s) {
    // inner conv + relu: y(i) for i in [2,528), stored at ybuf[c][i-2]
    for (int idx = tid; idx < CH * 526; idx += 256) {
      int c = idx / 526, ii = idx % 526;
      const float* cp = cur + ii;
      const float* rp = refs + ii;
      const float* wo = w1o + c * 5;
      const float* wr = w1r + c * 5;
      float y = cst[c];
      #pragma unroll
      for (int tp = 0; tp < 5; ++tp) y += wo[tp] * cp[tp] + wr[tp] * rp[tp];
      ybuf[c * 528 + ii] = fmaxf(y, 0.f);
    }
    __syncthreads();
    // outer conv + residual: valid i in [3,527)
    for (int i = 3 + tid; i < 527; i += 256) {
      float z = bias2;
      for (int c = 0; c < CH; ++c) {
        const float* yb = ybuf + c * 528 + (i - 3);
        const float* wc = w2s + c * 3;
        z += wc[0] * yb[0] + wc[1] * yb[1] + wc[2] * yb[2];
      }
      nxt[i] = cur[i] + 0.25f * z;
    }
    if (tid < 3) { nxt[tid] = cur[tid]; nxt[527 + tid] = cur[527 + tid]; }
    __syncthreads();
    float* t = cur; cur = nxt; nxt = t;
  }

  // write owned region and accumulate refined-pool partial
  float part = 0.f;
  for (int i = 9 + tid; i < 521; i += 256) {
    float v = cur[i];
    dout[(size_t)bk * T_ + t0 + (i - 9)] = v;
    part += v;
  }
  atomicAdd(&red[0], part);
  __syncthreads();
  if (tid == 0) atomicAdd(&ws[WS_RSUM + bk], red[0]);
}

// ---------------------------------------------------------------------------
// K4: refined_doa_logits = broadcast(mean_t out)
// ---------------------------------------------------------------------------
__global__ void k4_refpool(const float* __restrict__ ws, float* __restrict__ dout) {
  int tid = threadIdx.x;
  if (tid < B_ * KSPK) {
    float v = ws[WS_RSUM + tid] * (1.f / (float)T_);
    for (int j = 0; j < NBIN; ++j) dout[OUT2 + tid * NBIN + j] = v;
  }
}

// ---------------------------------------------------------------------------
extern "C" void kernel_launch(void* const* d_in, const int* in_sizes, int n_in,
                              void* d_out, int out_size, void* d_ws, size_t ws_size,
                              hipStream_t stream) {
  (void)in_sizes; (void)n_in; (void)out_size; (void)ws_size;
  const float* mix  = (const float*)d_in[0];
  const float* w1   = (const float*)d_in[1];
  const float* b1   = (const float*)d_in[2];
  const float* w2   = (const float*)d_in[3];
  const float* b2   = (const float*)d_in[4];
  const float* sepw = (const float*)d_in[5];
  const float* sepb = (const float*)d_in[6];
  const float* doaw = (const float*)d_in[7];
  const float* doab = (const float*)d_in[8];
  const float* embp = (const float*)d_in[9];
  const float* spw  = (const float*)d_in[10];
  const float* spb  = (const float*)d_in[11];
  const float* rw1  = (const float*)d_in[12];
  const float* rb1  = (const float*)d_in[13];
  const float* rw2  = (const float*)d_in[14];
  const float* rb2  = (const float*)d_in[15];
  float* ws   = (float*)d_ws;
  float* outp = (float*)d_out;

  const size_t sm1 = 65696 + (size_t)TT1 * KDIM * sizeof(half_t); // 147,616 B
  const size_t sm3 = 9984 + (size_t)CH * 528 * sizeof(float);     // 145,152 B
  static bool attr_done = false;
  // host-side attribute set; not a stream op, safe under graph capture
  hipFuncSetAttribute((const void*)k1_backbone,
                      hipFuncAttributeMaxDynamicSharedMemorySize, (int)sm1);
  hipFuncSetAttribute((const void*)k3_refine,
                      hipFuncAttributeMaxDynamicSharedMemorySize, (int)sm3);
  (void)attr_done;

  k0_init<<<1, 256, 0, stream>>>(ws);
  k1_backbone<<<B_ * NT1, 256, sm1, stream>>>(mix, w1, b1, w2, b2, sepw, sepb, ws);
  k2_head<<<1, 256, 0, stream>>>(doaw, doab, embp, spw, spb, rw1, rb1, ws, outp);
  k3_refine<<<B_ * KSPK * NT3, 256, sm3, stream>>>(mix, rw1, rw2, rb2, outp, ws);
  k4_refpool<<<1, 64, 0, stream>>>(ws, outp);
}
